// meta_model_local_sharp_36223754174520
// MI455X (gfx1250) — compile-verified, matrix operand order checked by reading, not run on hardware
//
#include <hip/hip_runtime.h>
#include <hip/hip_bf16.h>
#include <math.h>

// ---------------- problem constants ----------------
constexpr int LSTEPS = 10;
constexpr int NB  = 4;
constexpr int HH  = 256;
constexpr int WW  = 256;
constexpr int HWP = HH * WW;            // 65536 = 2^16
constexpr int HID = 100;
constexpr int HIDP = 112;               // 7 x 16 WMMA M tiles
constexpr int KC  = 18;                 // conv1 K = 2ch * 3 * 3
constexpr int KSZ = 25;                 // gaussian taps
constexpr float SIG = 4.0f;
constexpr float MUC = 0.02f;
constexpr float SLP = 0.01f;
constexpr int TW = 16;                  // res-block tile width  (WMMA N)
constexpr int TH = 8;                   // res-block tile height

typedef __attribute__((ext_vector_type(16))) _Float16 v16h;
typedef __attribute__((ext_vector_type(8)))  float    v8f;

#if __has_builtin(__builtin_amdgcn_global_load_async_to_lds_b32)
#define USE_ASYNC_LDS 1
typedef __attribute__((address_space(1))) int* as1_i32p;
typedef __attribute__((address_space(3))) int* as3_i32p;
#else
#define USE_ASYNC_LDS 0
#endif

// ---------------- helpers ----------------
__device__ __forceinline__ int clampi(int v, int lo, int hi) {
    return v < lo ? lo : (v > hi ? hi : v);
}

// bilinear sample with zeros padding (matches reference _deform corner masking)
__device__ __forceinline__ float bilin(const float* __restrict__ img, float gx, float gy) {
    float x0f = floorf(gx), y0f = floorf(gy);
    int ix0 = (int)x0f, iy0 = (int)y0f;
    float wx = gx - x0f, wy = gy - y0f;
    float v00 = 0.f, v01 = 0.f, v10 = 0.f, v11 = 0.f;
    {
        int iy = iy0, ix = ix0;
        if (ix >= 0 && ix < WW && iy >= 0 && iy < HH) v00 = img[iy * WW + ix];
        ix = ix0 + 1;
        if (ix >= 0 && ix < WW && iy >= 0 && iy < HH) v01 = img[iy * WW + ix];
        iy = iy0 + 1; ix = ix0;
        if (ix >= 0 && ix < WW && iy >= 0 && iy < HH) v10 = img[iy * WW + ix];
        ix = ix0 + 1;
        if (ix >= 0 && ix < WW && iy >= 0 && iy < HH) v11 = img[iy * WW + ix];
    }
    return v00 * (1.f - wx) * (1.f - wy) + v01 * wx * (1.f - wy)
         + v10 * (1.f - wx) * wy        + v11 * wx * wy;
}

// ---------------- init: res0 / rd0 / image0 / phi0 ----------------
__global__ void mmls_init_kernel(const float* __restrict__ source,
                                 const float* __restrict__ z0,
                                 float* __restrict__ res0_out,   // [B,HW] residuals slot 0
                                 float* __restrict__ rd0,        // [B,1,HW]
                                 float* __restrict__ img0,       // [B,HW]
                                 float* __restrict__ phi0)       // [B,2,HW] planar
{
    int idx = blockIdx.x * blockDim.x + threadIdx.x;
    if (idx >= NB * HWP) return;
    int x = idx & (WW - 1);
    int y = (idx >> 8) & (HH - 1);
    int b = idx >> 16;
    int pix = y * WW + x;
    float z = z0[pix];
    res0_out[idx] = z;
    rd0[b * HWP + pix] = z;
    img0[idx] = source[idx];
    phi0[(b * 2 + 0) * HWP + pix] = (float)x;
    phi0[(b * 2 + 1) * HWP + pix] = (float)y;
}

// ---------------- sobel (replicate pad, /8) -> grads slot [B,2,H,W] ----------------
__global__ void mmls_sobel_kernel(const float* __restrict__ img,
                                  float* __restrict__ grads)
{
    int idx = blockIdx.x * blockDim.x + threadIdx.x;
    if (idx >= NB * HWP) return;
    int x = idx & (WW - 1);
    int y = (idx >> 8) & (HH - 1);
    int b = idx >> 16;
    const float* p = img + b * HWP;
    int ym = clampi(y - 1, 0, HH - 1), yp = clampi(y + 1, 0, HH - 1);
    int xm = clampi(x - 1, 0, WW - 1), xp = clampi(x + 1, 0, WW - 1);
    float a00 = p[ym * WW + xm], a01 = p[ym * WW + x], a02 = p[ym * WW + xp];
    float a10 = p[y  * WW + xm],                        a12 = p[y  * WW + xp];
    float a20 = p[yp * WW + xm], a21 = p[yp * WW + x], a22 = p[yp * WW + xp];
    float dx = (-a00 + a02 - 2.f * a10 + 2.f * a12 - a20 + a22) * 0.125f;
    float dy = (-a00 - 2.f * a01 - a02 + a20 + 2.f * a21 + a22) * 0.125f;
    grads[(b * 2 + 0) * HWP + y * WW + x] = dx;
    grads[(b * 2 + 1) * HWP + y * WW + x] = dy;
}

__device__ __forceinline__ void make_gauss(float* g) {
    if (threadIdx.x == 0) {
        float s = 0.f, t[KSZ];
        for (int j = 0; j < KSZ; ++j) {
            float r = (float)j - (float)(KSZ - 1) * 0.5f;
            t[j] = __expf(-0.5f * (r / SIG) * (r / SIG));
            s += t[j];
        }
        float inv = 1.f / s;
        for (int j = 0; j < KSZ; ++j) g[j] = t[j] * inv;
    }
    __syncthreads();
}

// vertical blur of (-res * g2), zero border -> tmp [B,2,H,W]
__global__ void mmls_blur_v_kernel(const float* __restrict__ res,
                                   const float* __restrict__ g2,
                                   float* __restrict__ tmp)
{
    __shared__ float g[KSZ];
    make_gauss(g);
    int idx = blockIdx.x * blockDim.x + threadIdx.x;
    if (idx >= NB * 2 * HWP) return;
    int x = idx & (WW - 1);
    int y = (idx >> 8) & (HH - 1);
    int c = (idx >> 16) & 1;
    int b = idx >> 17;
    const float* rp = res + b * HWP;
    const float* gp = g2 + (b * 2 + c) * HWP;
    // hint the far end of the stencil into cache (global_prefetch_b8)
    __builtin_prefetch(&rp[clampi(y + KSZ / 2, 0, HH - 1) * WW + x], 0, 1);
    float s = 0.f;
    #pragma unroll
    for (int j = 0; j < KSZ; ++j) {
        int yy = y + j - (KSZ / 2);
        if (yy >= 0 && yy < HH) s += g[j] * (-rp[yy * WW + x] * gp[yy * WW + x]);
    }
    tmp[idx] = s;
}

// horizontal blur, zero border -> field slot [B,H,W,2]
__global__ void mmls_blur_h_kernel(const float* __restrict__ tmp,
                                   float* __restrict__ field)
{
    __shared__ float g[KSZ];
    make_gauss(g);
    int idx = blockIdx.x * blockDim.x + threadIdx.x;
    if (idx >= NB * 2 * HWP) return;
    int x = idx & (WW - 1);
    int y = (idx >> 8) & (HH - 1);
    int c = (idx >> 16) & 1;
    int b = idx >> 17;
    const float* tp = tmp + (b * 2 + c) * HWP + y * WW;
    __builtin_prefetch(&tp[clampi(x + KSZ / 2, 0, WW - 1)], 0, 1);
    float s = 0.f;
    #pragma unroll
    for (int j = 0; j < KSZ; ++j) {
        int xx = x + j - (KSZ / 2);
        if (xx >= 0 && xx < WW) s += g[j] * tp[xx];
    }
    field[((b * HH + y) * WW + x) * 2 + c] = s;
}

// ---------------- fused res-block via WMMA ----------------
// conv1 as implicit GEMM: M=112(ch pad), K=32(18 pad), N=16 pixels/row-tile,
// hidden tile kept entirely in LDS as f16; conv2 (K=900, M=1) on VALU f32.
__global__ __launch_bounds__(256) void mmls_resblock_kernel(
    const float* __restrict__ res_i,   // [B,HW]
    const float* __restrict__ image,   // [B,HW]
    const float* __restrict__ w1,      // [100,18] this step
    const float* __restrict__ b1,      // [100]
    const float* __restrict__ w2,      // [900] (ch*9 + ky*3 + kx)
    float* __restrict__ res_o)         // [B,HW] residuals slot i+1
{
    // input halo, padded so all fragment reads are unconditional:
    //   plane 2 is all-zero (serves K >= 18), cols >= 20 are all-zero.
    __shared__ _Float16 sh_in[3][TH + 4][36];
    __shared__ _Float16 sh_w1[HIDP][32];             // A matrix, K-padded
    __shared__ float    sh_b1[HIDP];
    __shared__ float    sh_w2[HID * 9];
    __shared__ _Float16 sh_h[TH + 2][TW + 2][HIDP];  // hidden halo 10x18x112
    __shared__ float    sh_part[TW * TH];            // conv2 partial sums
#if USE_ASYNC_LDS
    __shared__ float    sh_inf[2][TH + 4][20];       // async-DMA landing (f32)
#endif

    const int b   = blockIdx.z;
    const int tx0 = blockIdx.x * TW;
    const int ty0 = blockIdx.y * TH;
    const int t   = threadIdx.x;

    // ---- stage weights ----
    for (int idx = t; idx < HIDP * 32; idx += 256) {
        int m = idx >> 5, k = idx & 31;
        float v = (m < HID && k < KC) ? w1[m * KC + k] : 0.f;
        sh_w1[m][k] = (_Float16)v;
    }
    for (int idx = t; idx < HIDP; idx += 256)
        sh_b1[idx] = (idx < HID) ? b1[idx] : 0.f;
    for (int idx = t; idx < HID * 9; idx += 256)
        sh_w2[idx] = w2[idx];

    // ---- stage input halo ----
    for (int idx = t; idx < 3 * (TH + 4) * 36; idx += 256)
        (&sh_in[0][0][0])[idx] = (_Float16)0.f;

    const float* src0 = res_i + (size_t)b * HWP;
    const float* src1 = image + (size_t)b * HWP;

#if USE_ASYNC_LDS
    for (int idx = t; idx < 2 * (TH + 4) * 20; idx += 256)
        (&sh_inf[0][0][0])[idx] = 0.f;
    __syncthreads();
    // global -> LDS DMA (ASYNCcnt), no VGPR round-trip; OOB lanes keep zeros
    for (int idx = t; idx < 2 * (TH + 4) * 20; idx += 256) {
        int c = idx / ((TH + 4) * 20);
        int r = (idx / 20) % (TH + 4);
        int col = idx % 20;
        int y = ty0 - 2 + r, x = tx0 - 2 + col;
        if (y >= 0 && y < HH && x >= 0 && x < WW) {
            float* gp = const_cast<float*>((c == 0 ? src0 : src1) + y * WW + x);
            float* lp = &sh_inf[c][r][col];
            __builtin_amdgcn_global_load_async_to_lds_b32(
                (as1_i32p)gp, (as3_i32p)lp, 0, 0);
        }
    }
    asm volatile("s_wait_asynccnt 0" ::: "memory");
    __syncthreads();
    for (int idx = t; idx < 2 * (TH + 4) * 20; idx += 256) {
        int c = idx / ((TH + 4) * 20);
        int r = (idx / 20) % (TH + 4);
        int col = idx % 20;
        sh_in[c][r][col] = (_Float16)sh_inf[c][r][col];
    }
#else
    __syncthreads();
    for (int idx = t; idx < 2 * (TH + 4) * 20; idx += 256) {
        int c = idx / ((TH + 4) * 20);
        int r = (idx / 20) % (TH + 4);
        int col = idx % 20;
        int y = ty0 - 2 + r, x = tx0 - 2 + col;
        float v = 0.f;
        if (y >= 0 && y < HH && x >= 0 && x < WW)
            v = (c == 0 ? src0 : src1)[y * WW + x];
        sh_in[c][r][col] = (_Float16)v;
    }
#endif
    __syncthreads();

    const int wave   = t >> 5;
    const int lane   = t & 31;
    const int laneLo = lane & 15;
    const int laneHi = lane >> 4;

    // A fragments are invariant across N-tiles: load the 7 of them once.
    v16h afr[HIDP / 16];
    #pragma unroll
    for (int mt = 0; mt < HIDP / 16; ++mt) {
        #pragma unroll
        for (int e = 0; e < 16; ++e) {
            int K = ((e & 8) << 1) + (e & 7) + (laneHi << 3);
            afr[mt][e] = sh_w1[mt * 16 + laneLo][K];
        }
    }

    // 20 N-tiles: 10 hidden rows x 2 column-tiles (cols 0..15 / 16..31 rel)
    for (int nt = wave; nt < (TH + 2) * 2; nt += 8) {
        const int hr  = nt >> 1;
        const int hc0 = (nt & 1) << 4;
        const int hc  = hc0 + laneLo;

        // B fragment: 32(K) x 16(N), per 7.12.2 16-bit layout; all reads in-range
        v16h bfrag;
        #pragma unroll
        for (int e = 0; e < 16; ++e) {
            int K = ((e & 8) << 1) + (e & 7) + (laneHi << 3);
            int c  = (K >= KC) ? 2 : K / 9;     // plane 2 == zeros
            int rm = K % 9;
            bfrag[e] = sh_in[c][hr + rm / 3][hc + rm % 3];
        }

        const int ay = ty0 - 1 + hr;
        const int ax = tx0 - 1 + hc;
        const bool inimg = (ay >= 0 && ay < HH && ax >= 0 && ax < WW);

        #pragma unroll
        for (int mt = 0; mt < HIDP / 16; ++mt) {
            v8f acc = {0.f, 0.f, 0.f, 0.f, 0.f, 0.f, 0.f, 0.f};
            acc = __builtin_amdgcn_wmma_f32_16x16x32_f16(
                false, afr[mt], false, bfrag, (short)0, acc, false, false);
            #pragma unroll
            for (int r = 0; r < 8; ++r) {
                int ch = mt * 16 + r + (laneHi << 3);   // D: lanes>=16 hold M=8..15
                float v = acc[r] + sh_b1[ch];
                v = v >= 0.f ? v : SLP * v;
                if (!inimg) v = 0.f;                    // conv2 zero-pads hidden
                if (hc < TW + 2) sh_h[hr][hc][ch] = (_Float16)v;
            }
        }
    }
    __syncthreads();

    // conv2: 100ch x 3x3; 2 threads per pixel, 50 channels each, LDS reduce
    {
        int p = t >> 1;                 // output pixel 0..127
        int half = t & 1;               // channel half
        int ox = p & (TW - 1), oy = p >> 4;
        int ch0 = half * (HID / 2);
        float sum = 0.f;
        #pragma unroll
        for (int ky = 0; ky < 3; ++ky) {
            #pragma unroll
            for (int kx = 0; kx < 3; ++kx) {
                const _Float16* hp = &sh_h[oy + ky][ox + kx][ch0];
                const float* wp = &sh_w2[ch0 * 9 + ky * 3 + kx];
                for (int ch = 0; ch < HID / 2; ++ch)
                    sum += wp[ch * 9] * (float)hp[ch];
            }
        }
        if (half) sh_part[p] = sum;
        __syncthreads();
        if (!half) {
            float tot = sum + sh_part[p];
            int gi = b * HWP + (ty0 + oy) * WW + (tx0 + ox);
            res_o[gi] = res_i[gi] + tot * (1.f / (float)LSTEPS);
        }
    }
}

// ---------------- warp rd stack with deformation = id - field/L, append res_new ----------------
__global__ void mmls_warp_rd_kernel(const float* __restrict__ rd_in,   // [B,nc,HW]
                                    const float* __restrict__ field,   // [B,H,W,2]
                                    const float* __restrict__ res_new, // [B,HW]
                                    float* __restrict__ rd_out,        // [B,nc+1,HW]
                                    int nc)
{
    int idx = blockIdx.x * blockDim.x + threadIdx.x;
    int total = NB * (nc + 1) * HWP;
    if (idx >= total) return;
    int x = idx & (WW - 1);
    int y = (idx >> 8) & (HH - 1);
    int bc = idx >> 16;
    int c = bc % (nc + 1);
    int b = bc / (nc + 1);
    int pix = y * WW + x;
    if (c == nc) { rd_out[idx] = res_new[b * HWP + pix]; return; }
    const float* f = field + ((size_t)b * HWP + pix) * 2;
    float gx = (float)x - f[0] * (1.f / (float)LSTEPS);
    float gy = (float)y - f[1] * (1.f / (float)LSTEPS);
    rd_out[idx] = bilin(rd_in + ((size_t)b * nc + c) * HWP, gx, gy);
}

// ---------------- warp phi (planar [B,2,H,W]) ----------------
__global__ void mmls_warp_phi_kernel(const float* __restrict__ phi_in,
                                     const float* __restrict__ field,
                                     float* __restrict__ phi_out)
{
    int idx = blockIdx.x * blockDim.x + threadIdx.x;
    if (idx >= NB * 2 * HWP) return;
    int x = idx & (WW - 1);
    int y = (idx >> 8) & (HH - 1);
    int c = (idx >> 16) & 1;
    int b = idx >> 17;
    int pix = y * WW + x;
    const float* f = field + ((size_t)b * HWP + pix) * 2;
    float gx = (float)x - f[0] * (1.f / (float)LSTEPS);
    float gy = (float)y - f[1] * (1.f / (float)LSTEPS);
    phi_out[idx] = bilin(phi_in + (size_t)(b * 2 + c) * HWP, gx, gy);
}

// ---------------- image = warp(src,phi) + sum(rd[:,1:]) * mu^2/L * warp(seg,phi) ----------------
__global__ void mmls_compose_kernel(const float* __restrict__ source,
                                    const float* __restrict__ seg,
                                    const float* __restrict__ phi,    // [B,2,HW]
                                    const float* __restrict__ rd,     // [B,nc_rd,HW]
                                    int nc_rd,
                                    float* __restrict__ image_out)
{
    int idx = blockIdx.x * blockDim.x + threadIdx.x;
    if (idx >= NB * HWP) return;
    int x = idx & (WW - 1);
    int y = (idx >> 8) & (HH - 1);
    int b = idx >> 16;
    int pix = y * WW + x;
    float gx = phi[(b * 2 + 0) * HWP + pix];
    float gy = phi[(b * 2 + 1) * HWP + pix];
    float s = bilin(source + (size_t)b * HWP, gx, gy);
    float m = bilin(seg    + (size_t)b * HWP, gx, gy);
    float rsum = 0.f;
    for (int c = 1; c < nc_rd; ++c)
        rsum += rd[((size_t)b * nc_rd + c) * HWP + pix];
    image_out[idx] = s + rsum * (MUC * MUC / (float)LSTEPS) * m;
}

// ---------------- driver ----------------
extern "C" void kernel_launch(void* const* d_in, const int* in_sizes, int n_in,
                              void* d_out, int out_size, void* d_ws, size_t ws_size,
                              hipStream_t stream) {
    (void)in_sizes; (void)n_in; (void)out_size; (void)ws_size;
    const float* source = (const float*)d_in[0];
    const float* seg    = (const float*)d_in[1];
    const float* z0     = (const float*)d_in[2];
    const float* w1     = (const float*)d_in[3];   // [L,100,2,3,3]
    const float* b1     = (const float*)d_in[4];   // [L,100]
    const float* w2     = (const float*)d_in[5];   // [L,1,100,3,3]

    float* out = (float*)d_out;
    float* image_out  = out;                                   // B*HW
    float* fields_out = out + (size_t)NB * HWP;                // L*B*HW*2
    float* res_out    = fields_out + (size_t)LSTEPS * NB * HWP * 2;  // (L+1)*B*HW
    float* grads_out  = res_out + (size_t)(LSTEPS + 1) * NB * HWP;   // L*B*2*HW

    float* ws = (float*)d_ws;
    const size_t RD = (size_t)NB * (LSTEPS + 1) * HWP;
    const size_t PH = (size_t)NB * 2 * HWP;
    float* rdbuf[2]  = { ws, ws + RD };
    float* phibuf[2] = { ws + 2 * RD, ws + 2 * RD + PH };
    float* tmpb      = ws + 2 * RD + 2 * PH;
    float* imgbuf[2] = { tmpb + PH, tmpb + PH + (size_t)NB * HWP };

    const int NPIX = NB * HWP;
    dim3 blk(256);

    mmls_init_kernel<<<(NPIX + 255) / 256, blk, 0, stream>>>(
        source, z0, res_out, rdbuf[0], imgbuf[0], phibuf[0]);

    for (int i = 0; i < LSTEPS; ++i) {
        const float* img_c = imgbuf[i & 1];
        float* img_n = imgbuf[(i + 1) & 1];
        const float* res_i = res_out + (size_t)i * NPIX;
        float* res_n = res_out + (size_t)(i + 1) * NPIX;
        float* gslot = grads_out + (size_t)i * NB * 2 * HWP;
        float* fslot = fields_out + (size_t)i * NB * HWP * 2;

        mmls_sobel_kernel<<<(NPIX + 255) / 256, blk, 0, stream>>>(img_c, gslot);
        mmls_blur_v_kernel<<<(2 * NPIX + 255) / 256, blk, 0, stream>>>(res_i, gslot, tmpb);
        mmls_blur_h_kernel<<<(2 * NPIX + 255) / 256, blk, 0, stream>>>(tmpb, fslot);

        mmls_resblock_kernel<<<dim3(WW / TW, HH / TH, NB), blk, 0, stream>>>(
            res_i, img_c,
            w1 + (size_t)i * HID * KC,
            b1 + (size_t)i * HID,
            w2 + (size_t)i * HID * 9,
            res_n);

        int nc = i + 1;                       // rd channels before warp/append
        int tot = NB * (nc + 1) * HWP;
        mmls_warp_rd_kernel<<<(tot + 255) / 256, blk, 0, stream>>>(
            rdbuf[i & 1], fslot, res_n, rdbuf[(i + 1) & 1], nc);
        mmls_warp_phi_kernel<<<(2 * NPIX + 255) / 256, blk, 0, stream>>>(
            phibuf[i & 1], fslot, phibuf[(i + 1) & 1]);
        mmls_compose_kernel<<<(NPIX + 255) / 256, blk, 0, stream>>>(
            source, seg, phibuf[(i + 1) & 1], rdbuf[(i + 1) & 1], nc + 1, img_n);
    }

    (void)hipMemcpyAsync(image_out, imgbuf[LSTEPS & 1], (size_t)NPIX * sizeof(float),
                         hipMemcpyDeviceToDevice, stream);
}
